// TransformerLayerTorch_69647189672021
// MI455X (gfx1250) — matrix operand order repaired and checked by reading, not compiled, not run on hardware
//
#include <hip/hip_runtime.h>
#include <cstdint>
#include <cstddef>

// ---------------------------------------------------------------------------
// Transformer layer for MI455X (gfx1250): bf16 WMMA everywhere, flash attention.
// B=4, S=2048, D=1024, H=16, dh=64, FF=4096.
// ---------------------------------------------------------------------------

#define D_MODEL 1024
#define N_HEADS 16
#define D_HEAD  64
#define D_FF    4096
#define BATCH   4
#define SEQ     2048
#define ROWS    (BATCH * SEQ)   // 8192
#define LN_EPS  1e-5f

typedef __attribute__((ext_vector_type(16))) __bf16       v16bf;
typedef __attribute__((ext_vector_type(8)))  float        v8f;
typedef __attribute__((ext_vector_type(4)))  unsigned int v4u;   // 16B chunk
typedef int v4i_vs __attribute__((vector_size(16)));             // builtin's pointee

// CDNA5 async global->LDS path (ASYNCcnt), guarded so absence falls back to
// synchronous b128 copies through VGPRs.
#if __has_builtin(__builtin_amdgcn_global_load_async_to_lds_b128) && \
    __has_builtin(__builtin_amdgcn_s_wait_asynccnt)
#define USE_ASYNC_LDS 1
#endif

// A/B fragment for v_wmma_f32_16x16x32_bf16: 16 bf16 = two 16B chunks per lane.
union FragAB {
    v16bf v;
    v4u   q[2];
};

__device__ __forceinline__ unsigned short f2bf(float f) {
    unsigned int u = __float_as_uint(f);
    unsigned int r = u + 0x7FFFu + ((u >> 16) & 1u);   // round-to-nearest-even
    return (unsigned short)(r >> 16);
}

__device__ __forceinline__ v8f wmma_bf16(const FragAB& a, const FragAB& b, v8f c) {
    return __builtin_amdgcn_wmma_f32_16x16x32_bf16(
        /*neg_a=*/false, a.v, /*neg_b=*/false, b.v,
        /*c_mod=*/(short)0, c, /*reuse_a=*/false, /*reuse_b=*/false);
}

// 16-byte global -> LDS copy; async (tracked by ASYNCcnt) when available.
__device__ __forceinline__ void cp16_g2l(const unsigned short* g, unsigned short* l) {
#ifdef USE_ASYNC_LDS
    __builtin_amdgcn_global_load_async_to_lds_b128(
        (__attribute__((address_space(1))) v4i_vs*)g,
        (__attribute__((address_space(3))) v4i_vs*)l, 0, 0);
#else
    *(v4u*)l = *(const v4u*)g;
#endif
}

__device__ __forceinline__ void wait_stage() {
#ifdef USE_ASYNC_LDS
    __builtin_amdgcn_s_wait_asynccnt(0);
#endif
}

// ---------------------------------------------------------------------------
// fp32 -> bf16 weight conversion
// ---------------------------------------------------------------------------
__global__ void cvt_bf16_kernel(const float* __restrict__ in,
                                unsigned short* __restrict__ out, int n) {
    int i = blockIdx.x * 256 + threadIdx.x;
    if (i < n) out[i] = f2bf(in[i]);
}

// ---------------------------------------------------------------------------
// Row LayerNorm (D=1024), fp32 in -> bf16 out. One 256-thread block per row.
// ---------------------------------------------------------------------------
__global__ void __launch_bounds__(256)
ln_bf16_kernel(const float* __restrict__ x,
               const float* __restrict__ gamma,
               const float* __restrict__ beta,
               unsigned short* __restrict__ out) {
    __shared__ float red0[256];
    __shared__ float red1[256];
    const int row = blockIdx.x;
    const float* xr = x + (size_t)row * D_MODEL;
    float v[4], s = 0.f, ss = 0.f;
#pragma unroll
    for (int i = 0; i < 4; ++i) {
        v[i] = xr[threadIdx.x + 256 * i];
        s  += v[i];
        ss += v[i] * v[i];
    }
    red0[threadIdx.x] = s;
    red1[threadIdx.x] = ss;
    __syncthreads();
    for (int off = 128; off > 0; off >>= 1) {
        if (threadIdx.x < off) {
            red0[threadIdx.x] += red0[threadIdx.x + off];
            red1[threadIdx.x] += red1[threadIdx.x + off];
        }
        __syncthreads();
    }
    const float mean = red0[0] * (1.f / D_MODEL);
    const float var  = red1[0] * (1.f / D_MODEL) - mean * mean;
    const float rstd = rsqrtf(var + LN_EPS);
#pragma unroll
    for (int i = 0; i < 4; ++i) {
        int c = threadIdx.x + 256 * i;
        float nv = (v[i] - mean) * rstd * gamma[c] + beta[c];
        out[(size_t)row * D_MODEL + c] = f2bf(nv);
    }
}

// ---------------------------------------------------------------------------
// Generic GEMM: C[M,N] = A[M,K] (bf16, row-major) x W[N,K]^T (bf16, row-major)
// Block tile 128x128, 8 wave32s arranged 4(m) x 2(n), wave tile 32x64,
// K-step 32, double-buffered LDS (async global->LDS on CDNA5), fp32 accumulate.
// Epilogues: 0 = bf16 store, 1 = transposed-V bf16 store ([B,H,dh,S]),
//            2 = +residual fp32 store, 3 = GELU -> bf16 store.
// ---------------------------------------------------------------------------
template <int EPI>
__global__ void __launch_bounds__(256)
gemm_bf16_wmma(const unsigned short* __restrict__ A,
               const unsigned short* __restrict__ W,
               int M, int N, int K,
               const float* __restrict__ resid,
               float* __restrict__ outf,
               unsigned short* __restrict__ outh) {
    constexpr int AS = 40;   // 32 + 8 bf16 pad (80B row, keeps 16B alignment)
    constexpr int WS = 40;
    __shared__ __align__(16) unsigned short Asm[2][128 * AS];
    __shared__ __align__(16) unsigned short Wsm[2][128 * WS];

    const int tid  = threadIdx.x;
    const int lane = tid & 31;
    const int wave = tid >> 5;
    const int lh   = lane >> 4;    // lane-half (0/1)
    const int l16  = lane & 15;
    const int wm   = (wave & 3) * 32;    // wave row offset in block tile
    const int wn   = (wave >> 2) * 64;   // wave col offset in block tile
    const int m0   = blockIdx.y * 128;
    const int n0   = blockIdx.x * 128;

    const v8f zro = {0.f, 0.f, 0.f, 0.f, 0.f, 0.f, 0.f, 0.f};
    v8f acc[2][4];
#pragma unroll
    for (int i = 0; i < 2; ++i)
#pragma unroll
        for (int j = 0; j < 4; ++j) acc[i][j] = zro;

    const int r0  = tid >> 2;          // 0..63
    const int seg = (tid & 3) * 8;     // 0,8,16,24 (bf16 units)

    // Stage one 128x32 A tile + 128x32 W tile into LDS buffer `buf`.
    auto stage = [&](int buf, int k0) {
        unsigned short* Ab = &Asm[buf][0];
        unsigned short* Wb = &Wsm[buf][0];
        cp16_g2l(&A[(size_t)(m0 + r0) * K + k0 + seg],      &Ab[r0 * AS + seg]);
        cp16_g2l(&A[(size_t)(m0 + r0 + 64) * K + k0 + seg], &Ab[(r0 + 64) * AS + seg]);
        cp16_g2l(&W[(size_t)(n0 + r0) * K + k0 + seg],      &Wb[r0 * WS + seg]);
        cp16_g2l(&W[(size_t)(n0 + r0 + 64) * K + k0 + seg], &Wb[(r0 + 64) * WS + seg]);
#ifndef USE_ASYNC_LDS
        if (k0 + 32 < K) {  // global_prefetch_b8 of the following K-tile
            __builtin_prefetch(&A[(size_t)(m0 + r0) * K + k0 + 32 + seg], 0, 1);
            __builtin_prefetch(&W[(size_t)(n0 + r0) * K + k0 + 32 + seg], 0, 1);
        }
#endif
    };

    int cur = 0;
    stage(cur, 0);

    for (int k0 = 0; k0 < K; k0 += 32) {
        wait_stage();      // own async copies into lds[cur] complete
        __syncthreads();   // everyone's copies visible; prior compute retired
        if (k0 + 32 < K) stage(cur ^ 1, k0 + 32);   // overlap with compute below

        const unsigned short* Ab = &Asm[cur][0];
        const unsigned short* Wb = &Wsm[cur][0];

        // A fragment (ISA 16-bit A layout): lane holds row M=l16,
        // K = lh*8 + {0..7} and lh*8 + {16..23}  -> two contiguous 16B reads.
        FragAB af[2], bfr[4];
#pragma unroll
        for (int i = 0; i < 2; ++i) {
            const int r = wm + i * 16 + l16;
            af[i].q[0] = *(const v4u*)&Ab[r * AS + lh * 8];
            af[i].q[1] = *(const v4u*)&Ab[r * AS + lh * 8 + 16];
        }
        // B fragment: col N=l16, K = lh*8 + {0..7} and lh*8 + {16..23} (matches A).
#pragma unroll
        for (int j = 0; j < 4; ++j) {
            const int c = wn + j * 16 + l16;
            bfr[j].q[0] = *(const v4u*)&Wb[c * WS + lh * 8];
            bfr[j].q[1] = *(const v4u*)&Wb[c * WS + lh * 8 + 16];
        }
#pragma unroll
        for (int i = 0; i < 2; ++i)
#pragma unroll
            for (int j = 0; j < 4; ++j)
                acc[i][j] = wmma_bf16(af[i], bfr[j], acc[i][j]);

        cur ^= 1;
    }

    // Epilogue. C layout: element r -> (M = r + 8*lh, N = l16).
#pragma unroll
    for (int i = 0; i < 2; ++i)
#pragma unroll
        for (int j = 0; j < 4; ++j)
#pragma unroll
            for (int r = 0; r < 8; ++r) {
                const int Mg = m0 + wm + i * 16 + r + lh * 8;
                const int Ng = n0 + wn + j * 16 + l16;
                const float v = acc[i][j][r];
                const size_t idx = (size_t)Mg * N + Ng;
                if (EPI == 0) {
                    outh[idx] = f2bf(v);
                } else if (EPI == 1) {
                    // V^T store: [B, H, dh, S] so attention P@V reads contiguously.
                    const int b = Mg >> 11, s = Mg & 2047;
                    const int h = Ng >> 6,  e = Ng & 63;
                    outh[((size_t)(b * N_HEADS + h) * D_HEAD + e) * SEQ + s] = f2bf(v);
                } else if (EPI == 2) {
                    outf[idx] = v + resid[idx];
                } else {
                    const float c3 = v * v * v;
                    const float g =
                        0.5f * v * (1.f + tanhf(0.7978845608f * (v + 0.044715f * c3)));
                    outh[idx] = f2bf(g);
                }
            }
}

// ---------------------------------------------------------------------------
// Flash attention. Grid: (S/256, B*H). 8 waves; each wave owns 32 query rows
// (two 16-row sub-blocks) so every K/V fragment feeds two WMMAs. Q/K fragments
// loaded straight from global in WMMA A/B layouts; V read from the
// pre-transposed [B,H,dh,S] buffer. Scores never touch HBM.
// ---------------------------------------------------------------------------
__global__ void __launch_bounds__(256)
attn_kernel(const unsigned short* __restrict__ Q,   // [ROWS, D_MODEL] bf16
            const unsigned short* __restrict__ Kb,  // [ROWS, D_MODEL] bf16
            const unsigned short* __restrict__ Vt,  // [B, H, dh, S]   bf16
            unsigned short* __restrict__ O) {       // [ROWS, D_MODEL] bf16
    __shared__ __align__(16) unsigned short Pst[8][32 * 32];  // per-wave P staging

    const int lane = threadIdx.x & 31;
    const int wave = threadIdx.x >> 5;
    const int lh   = lane >> 4;
    const int l16  = lane & 15;
    const int b    = blockIdx.y >> 4;
    const int h    = blockIdx.y & 15;
    const int q0   = blockIdx.x * 256 + wave * 32;        // within-seq base row
    const size_t qrowg = (size_t)b * SEQ + q0;            // global row base

    // Q fragments (A layout) directly from global: rows l16, 16B chunks.
    const unsigned short* qbase = Q + qrowg * D_MODEL + h * D_HEAD;
    FragAB qf[2][2];
#pragma unroll
    for (int qs = 0; qs < 2; ++qs)
#pragma unroll
        for (int c = 0; c < 2; ++c) {
            const unsigned short* p =
                qbase + (size_t)(qs * 16 + l16) * D_MODEL + c * 32 + lh * 8;
            qf[qs][c].q[0] = *(const v4u*)p;
            qf[qs][c].q[1] = *(const v4u*)(p + 16);
        }

    const v8f zro = {0.f, 0.f, 0.f, 0.f, 0.f, 0.f, 0.f, 0.f};
    v8f ov[2][4];                      // two 16x64 output accumulators
    float mrow[2][8], lrow[2][8];
#pragma unroll
    for (int qs = 0; qs < 2; ++qs) {
#pragma unroll
        for (int dc = 0; dc < 4; ++dc) ov[qs][dc] = zro;
#pragma unroll
        for (int r = 0; r < 8; ++r) { mrow[qs][r] = -1e30f; lrow[qs][r] = 0.f; }
    }

    const unsigned short* vb = Vt + (size_t)(b * N_HEADS + h) * D_HEAD * SEQ;
    const int ntile = (blockIdx.x + 1) * 8;   // causal: 32-key tiles needed

    for (int kt = 0; kt < ntile; ++kt) {
        const int kb0 = kt * 32;

        // ---- scores: S(32x32) = Q(32x64) . K^T; K frags reused by both qs ----
        v8f sc[2][2] = {{zro, zro}, {zro, zro}};
#pragma unroll
        for (int kc = 0; kc < 2; ++kc)        // key chunk (N)
#pragma unroll
            for (int dc = 0; dc < 2; ++dc) {  // head-dim chunk (K)
                FragAB bfr;  // B[k][n] = K[key=n][dh=k] -> contiguous per key row
                const unsigned short* p =
                    Kb + ((size_t)b * SEQ + kb0 + kc * 16 + l16) * D_MODEL +
                    h * D_HEAD + dc * 32 + lh * 8;
                bfr.q[0] = *(const v4u*)p;
                bfr.q[1] = *(const v4u*)(p + 16);
#pragma unroll
                for (int qs = 0; qs < 2; ++qs)
                    sc[qs][kc] = wmma_bf16(qf[qs][dc], bfr, sc[qs][kc]);
            }

        // ---- scale + causal mask + online softmax per 16-row sub-block ----
#pragma unroll
        for (int qs = 0; qs < 2; ++qs) {
            const int qb0 = q0 + qs * 16;
            const bool needmask = (kb0 + 31) > qb0;
            float mnew[8];
#pragma unroll
            for (int r = 0; r < 8; ++r) {
                const int qr = qb0 + r + lh * 8;
                float mx = -1e30f;
#pragma unroll
                for (int kc = 0; kc < 2; ++kc) {
                    float s = sc[qs][kc][r] * 0.125f;
                    const int key = kb0 + kc * 16 + l16;
                    if (needmask && key > qr) s = -1e30f;
                    sc[qs][kc][r] = s;
                    mx = fmaxf(mx, s);
                }
                for (int off = 1; off < 16; off <<= 1)
                    mx = fmaxf(mx, __shfl_xor(mx, off, 32));
                mnew[r] = fmaxf(mrow[qs][r], mx);
            }
#pragma unroll
            for (int r = 0; r < 8; ++r) {
                const float corr = __expf(mrow[qs][r] - mnew[r]);
                float psum = 0.f;
#pragma unroll
                for (int kc = 0; kc < 2; ++kc) {
                    const float p = __expf(sc[qs][kc][r] - mnew[r]);
                    sc[qs][kc][r] = p;
                    psum += p;
                }
                for (int off = 1; off < 16; off <<= 1)
                    psum += __shfl_xor(psum, off, 32);
                lrow[qs][r] = lrow[qs][r] * corr + psum;
                mrow[qs][r] = mnew[r];
#pragma unroll
                for (int dc = 0; dc < 4; ++dc)
                    ov[qs][dc][r] = ov[qs][dc][r] * corr;
            }
        }

        // ---- C-layout -> A-layout for P via per-wave LDS staging ----
#pragma unroll
        for (int qs = 0; qs < 2; ++qs)
#pragma unroll
            for (int kc = 0; kc < 2; ++kc)
#pragma unroll
                for (int r = 0; r < 8; ++r)
                    Pst[wave][qs * 512 + (r + lh * 8) * 32 + kc * 16 + l16] =
                        f2bf(sc[qs][kc][r]);
#if __has_builtin(__builtin_amdgcn_wave_barrier)
        __builtin_amdgcn_wave_barrier();   // wave-private region: scheduling fence
#endif
        FragAB pf[2];
#pragma unroll
        for (int qs = 0; qs < 2; ++qs) {
            const unsigned short* p = &Pst[wave][qs * 512 + l16 * 32 + lh * 8];
            pf[qs].q[0] = *(const v4u*)p;
            pf[qs].q[1] = *(const v4u*)(p + 16);
        }

        // ---- ov(32x64) += P(32x32) . V(32x64); V frags reused by both qs ----
#pragma unroll
        for (int dc = 0; dc < 4; ++dc) {
            FragAB bfr;   // B from V^T: contiguous along sequence
            const unsigned short* p =
                vb + (size_t)(dc * 16 + l16) * SEQ + kb0 + lh * 8;
            bfr.q[0] = *(const v4u*)p;
            bfr.q[1] = *(const v4u*)(p + 16);
#pragma unroll
            for (int qs = 0; qs < 2; ++qs)
                ov[qs][dc] = wmma_bf16(pf[qs], bfr, ov[qs][dc]);
        }
    }

    // Normalize and write head-concat output [row, h*64 + e] as bf16.
#pragma unroll
    for (int qs = 0; qs < 2; ++qs)
#pragma unroll
        for (int dc = 0; dc < 4; ++dc)
#pragma unroll
            for (int r = 0; r < 8; ++r) {
                const size_t row = qrowg + qs * 16 + r + lh * 8;
                const float v = ov[qs][dc][r] / lrow[qs][r];
                O[row * D_MODEL + h * D_HEAD + dc * 16 + l16] = f2bf(v);
            }
}

// ---------------------------------------------------------------------------
// Host-side orchestration
// ---------------------------------------------------------------------------
extern "C" void kernel_launch(void* const* d_in, const int* in_sizes, int n_in,
                              void* d_out, int out_size, void* d_ws, size_t ws_size,
                              hipStream_t stream) {
    const float* x      = (const float*)d_in[0];
    const float* gamma1 = (const float*)d_in[1];
    const float* beta1  = (const float*)d_in[2];
    const float* wq     = (const float*)d_in[3];
    const float* wk     = (const float*)d_in[4];
    const float* wv     = (const float*)d_in[5];
    const float* wo     = (const float*)d_in[6];
    const float* gamma2 = (const float*)d_in[7];
    const float* beta2  = (const float*)d_in[8];
    const float* fc1    = (const float*)d_in[9];
    const float* fc2    = (const float*)d_in[10];
    float* out = (float*)d_out;

    // Workspace layout (bytes)
    char* ws = (char*)d_ws;
    size_t off = 0;
    auto alloc = [&](size_t bytes) { void* p = ws + off; off += (bytes + 255) & ~size_t(255); return p; };
    unsigned short* n1  = (unsigned short*)alloc((size_t)ROWS * D_MODEL * 2);  // 16MB
    unsigned short* qb  = (unsigned short*)alloc((size_t)ROWS * D_MODEL * 2);
    unsigned short* kb  = (unsigned short*)alloc((size_t)ROWS * D_MODEL * 2);
    unsigned short* vT  = (unsigned short*)alloc((size_t)ROWS * D_MODEL * 2);
    unsigned short* cc  = (unsigned short*)alloc((size_t)ROWS * D_MODEL * 2);
    float*          xr  = (float*)alloc((size_t)ROWS * D_MODEL * 4);           // 32MB
    unsigned short* n2  = (unsigned short*)alloc((size_t)ROWS * D_MODEL * 2);
    unsigned short* hid = (unsigned short*)alloc((size_t)ROWS * D_FF * 2);     // 64MB
    unsigned short* wqb = (unsigned short*)alloc((size_t)D_MODEL * D_MODEL * 2);
    unsigned short* wkb = (unsigned short*)alloc((size_t)D_MODEL * D_MODEL * 2);
    unsigned short* wvb = (unsigned short*)alloc((size_t)D_MODEL * D_MODEL * 2);
    unsigned short* wob = (unsigned short*)alloc((size_t)D_MODEL * D_MODEL * 2);
    unsigned short* f1b = (unsigned short*)alloc((size_t)D_FF * D_MODEL * 2);
    unsigned short* f2b = (unsigned short*)alloc((size_t)D_MODEL * D_FF * 2);
    (void)ws_size; (void)in_sizes; (void)n_in; (void)out_size;

    // 1. Weight conversions fp32 -> bf16
    auto cvt = [&](const float* src, unsigned short* dst, int n) {
        cvt_bf16_kernel<<<(n + 255) / 256, 256, 0, stream>>>(src, dst, n);
    };
    cvt(wq,  wqb, D_MODEL * D_MODEL);  // [H,dh,D] contiguous == [1024,1024]
    cvt(wk,  wkb, D_MODEL * D_MODEL);
    cvt(wv,  wvb, D_MODEL * D_MODEL);
    cvt(wo,  wob, D_MODEL * D_MODEL);
    cvt(fc1, f1b, D_FF * D_MODEL);
    cvt(fc2, f2b, D_MODEL * D_FF);

    // 2. LayerNorm1
    ln_bf16_kernel<<<ROWS, 256, 0, stream>>>(x, gamma1, beta1, n1);

    // 3. QKV projections (V stored transposed for attention)
    dim3 gQKV(D_MODEL / 128, ROWS / 128);
    gemm_bf16_wmma<0><<<gQKV, 256, 0, stream>>>(n1, wqb, ROWS, D_MODEL, D_MODEL,
                                                nullptr, nullptr, qb);
    gemm_bf16_wmma<0><<<gQKV, 256, 0, stream>>>(n1, wkb, ROWS, D_MODEL, D_MODEL,
                                                nullptr, nullptr, kb);
    gemm_bf16_wmma<1><<<gQKV, 256, 0, stream>>>(n1, wvb, ROWS, D_MODEL, D_MODEL,
                                                nullptr, nullptr, vT);

    // 4. Flash attention -> head-concat bf16
    attn_kernel<<<dim3(SEQ / 256, BATCH * N_HEADS), 256, 0, stream>>>(qb, kb, vT, cc);

    // 5. Output projection + residual -> x_resid (fp32)
    gemm_bf16_wmma<2><<<gQKV, 256, 0, stream>>>(cc, wob, ROWS, D_MODEL, D_MODEL,
                                                x, xr, nullptr);

    // 6. LayerNorm2
    ln_bf16_kernel<<<ROWS, 256, 0, stream>>>(xr, gamma2, beta2, n2);

    // 7. FC1 + GELU -> hidden (bf16)
    gemm_bf16_wmma<3><<<dim3(D_FF / 128, ROWS / 128), 256, 0, stream>>>(
        n2, f1b, ROWS, D_FF, D_MODEL, nullptr, nullptr, hid);

    // 8. FC2 + residual -> output (fp32)
    gemm_bf16_wmma<2><<<gQKV, 256, 0, stream>>>(hid, f2b, ROWS, D_MODEL, D_FF,
                                                xr, out, nullptr);
}
